// SparseMoE_24764781428918
// MI455X (gfx1250) — matrix operand order repaired and checked by reading, not compiled, not run on hardware
//
#include <hip/hip_runtime.h>

#define T_TOKENS   2048
#define DMODEL     768
#define DFF        2048
#define NEXPERTS   8
#define TM         32        // token tile = 2 WMMA M-tiles

typedef __attribute__((ext_vector_type(16))) __bf16 v16bf;
typedef __attribute__((ext_vector_type(8)))  float  v8f;
typedef __attribute__((ext_vector_type(8)))  unsigned int v8u;   // 32B fragment

union Frag { v8u u; v16bf v; };

// f32 -> bf16 bits, round-to-nearest-even
__device__ __forceinline__ unsigned int f2bf(float f) {
  union { float f; unsigned int u; } c; c.f = f;
  return (c.u + 0x7fffu + ((c.u >> 16) & 1u)) >> 16;
}

// workspace layout (bytes); total ~75.2 MB
#define WS_COUNTS 0
#define WS_TLIST  256
#define WS_TWT    (WS_TLIST + NEXPERTS * T_TOKENS * 4)           //  65792
#define WS_XBF    (WS_TWT   + NEXPERTS * T_TOKENS * 4)           // 131328
#define WS_WG     (WS_XBF   + (size_t)T_TOKENS * DMODEL * 2)     // 3277056
#define WS_WU     (WS_WG    + (size_t)NEXPERTS * DMODEL * DFF * 2)
#define WS_WD     (WS_WU    + (size_t)NEXPERTS * DMODEL * DFF * 2)

// ---------------------------------------------------------------------------
// zero output (incl. loss scalar) + expert counters
// ---------------------------------------------------------------------------
__global__ void moe_init(float* __restrict__ out, int out_n, int* __restrict__ counts) {
  int i = blockIdx.x * 256 + threadIdx.x;
  if (i < out_n) out[i] = 0.0f;
  if (i < NEXPERTS) counts[i] = 0;
}

// ---------------------------------------------------------------------------
// x (f32 row-major) -> bf16 row-major
// ---------------------------------------------------------------------------
__global__ __launch_bounds__(256) void convert_x(const float* __restrict__ x,
                                                 unsigned short* __restrict__ xbf) {
  int i = blockIdx.x * 256 + threadIdx.x;            // < T*DMODEL/4
  float4 v = ((const float4*)x)[i];
  uint2 p;
  p.x = f2bf(v.x) | (f2bf(v.y) << 16);
  p.y = f2bf(v.z) | (f2bf(v.w) << 16);
  ((uint2*)xbf)[i] = p;
}

// ---------------------------------------------------------------------------
// w_gate / w_up [E][D][F] f32 -> bf16 B-fragments:
//   frag = ((e*128 + f/16)*24 + d/32),  lane = 16*((d%32)/16) + f%16,  j = d%16
// Each thread: 4 consecutive d (one f) -> one contiguous 8B store.
// ---------------------------------------------------------------------------
__global__ __launch_bounds__(256) void convert_wgu(
    const float* __restrict__ wg, const float* __restrict__ wu,
    unsigned short* __restrict__ wgB, unsigned short* __restrict__ wuB) {
  int i = blockIdx.x * 256 + threadIdx.x;            // < E*(D/4)*F
  int f  = i & (DFF - 1);
  int r2 = i >> 11;
  int dg = r2 % (DMODEL / 4);
  int e  = r2 / (DMODEL / 4);
  size_t src = ((size_t)e * DMODEL + dg * 4) * DFF + f;
  int d0 = dg * 4, kt = d0 >> 5, kk0 = d0 & 31;
  int lane = 16 * (kk0 >> 4) + (f & 15);
  size_t dst = (((size_t)(e * (DFF / 16) + (f >> 4)) * (DMODEL / 32) + kt) * 512
                + lane * 16 + (kk0 & 15)) >> 2;      // uint2 index
  float g0 = wg[src], g1 = wg[src + DFF], g2 = wg[src + 2 * DFF], g3 = wg[src + 3 * DFF];
  float u0 = wu[src], u1 = wu[src + DFF], u2 = wu[src + 2 * DFF], u3 = wu[src + 3 * DFF];
  uint2 pg, pu;
  pg.x = f2bf(g0) | (f2bf(g1) << 16); pg.y = f2bf(g2) | (f2bf(g3) << 16);
  pu.x = f2bf(u0) | (f2bf(u1) << 16); pu.y = f2bf(u2) | (f2bf(u3) << 16);
  ((uint2*)wgB)[dst] = pg;
  ((uint2*)wuB)[dst] = pu;
}

// ---------------------------------------------------------------------------
// w_down [E][F][D] f32 -> bf16 B-fragments (K = F, N = D):
//   frag = ((e*48 + d/16)*64 + f/32),  lane = 16*((f%32)/16) + d%16,  j = f%16
// ---------------------------------------------------------------------------
__global__ __launch_bounds__(256) void convert_wd(
    const float* __restrict__ wd, unsigned short* __restrict__ wdB) {
  int i = blockIdx.x * 256 + threadIdx.x;            // < E*(F/4)*D
  int d  = i % DMODEL;
  int r2 = i / DMODEL;
  int fg = r2 & (DFF / 4 - 1);
  int e  = r2 >> 9;
  size_t src = ((size_t)e * DFF + fg * 4) * DMODEL + d;
  int f0 = fg * 4, kt = f0 >> 5, kk0 = f0 & 31;
  int lane = 16 * (kk0 >> 4) + (d & 15);
  size_t dst = (((size_t)(e * (DMODEL / 16) + (d >> 4)) * (DFF / 32) + kt) * 512
                + lane * 16 + (kk0 & 15)) >> 2;
  float a0 = wd[src], a1 = wd[src + DMODEL], a2 = wd[src + 2 * DMODEL], a3 = wd[src + 3 * DMODEL];
  uint2 p;
  p.x = f2bf(a0) | (f2bf(a1) << 16); p.y = f2bf(a2) | (f2bf(a3) << 16);
  ((uint2*)wdB)[dst] = p;
}

// ---------------------------------------------------------------------------
// router: one wave32 per token; top-2 softmax gating; compacted expert lists
// ---------------------------------------------------------------------------
__global__ __launch_bounds__(256) void moe_router(
    const float* __restrict__ x, const float* __restrict__ gate_w,
    int* __restrict__ counts, int* __restrict__ tlist, float* __restrict__ twt) {
  __shared__ float sGW[DMODEL * NEXPERTS];
  for (int i = threadIdx.x; i < DMODEL * NEXPERTS; i += 256) sGW[i] = gate_w[i];
  __syncthreads();

  const int lane = threadIdx.x & 31;
  const int t    = blockIdx.x * 8 + (threadIdx.x >> 5);

  float acc[NEXPERTS];
#pragma unroll
  for (int e = 0; e < NEXPERTS; ++e) acc[e] = 0.0f;
  for (int d = lane; d < DMODEL; d += 32) {
    float xv = x[(size_t)t * DMODEL + d];
#pragma unroll
    for (int e = 0; e < NEXPERTS; ++e) acc[e] += xv * sGW[d * NEXPERTS + e];
  }
#pragma unroll
  for (int e = 0; e < NEXPERTS; ++e)
    for (int off = 16; off > 0; off >>= 1)
      acc[e] += __shfl_down(acc[e], off, 32);

  if (lane == 0) {
    int i1 = 0;
    for (int e = 1; e < NEXPERTS; ++e) if (acc[e] > acc[i1]) i1 = e;
    int i2 = (i1 == 0) ? 1 : 0;
    for (int e = 0; e < NEXPERTS; ++e) if (e != i1 && acc[e] > acc[i2]) i2 = e;
    float m  = fmaxf(acc[i1], acc[i2]);
    float e1 = __expf(acc[i1] - m), e2 = __expf(acc[i2] - m);
    float inv = 1.0f / (e1 + e2);
    int p1 = atomicAdd(&counts[i1], 1);
    tlist[i1 * T_TOKENS + p1] = t; twt[i1 * T_TOKENS + p1] = e1 * inv;
    int p2 = atomicAdd(&counts[i2], 1);
    tlist[i2 * T_TOKENS + p2] = t; twt[i2 * T_TOKENS + p2] = e2 * inv;
  }
}

// ---------------------------------------------------------------------------
// grouped expert FFN. Block = 8 waves, 32-token tile (2 M-tiles).
// B operands: direct global->VGPR 32B fragment loads (L2-resident, pre-swizzled).
// A operands: 32B LDS fragment loads (x staged once; h written fragment-form).
// ---------------------------------------------------------------------------
__global__ __launch_bounds__(256, 1) void moe_expert_ffn(
    const unsigned short* __restrict__ xbf,
    const unsigned short* __restrict__ wgB,
    const unsigned short* __restrict__ wuB,
    const unsigned short* __restrict__ wdB,
    const int* __restrict__ counts, const int* __restrict__ tlist,
    const float* __restrict__ twt, float* __restrict__ out) {
  const int e   = blockIdx.y;
  const int n_e = counts[e];
  const int t0  = blockIdx.x * TM;
  if (t0 >= n_e) return;                             // uniform

  __shared__ __align__(32) unsigned short sXf[DMODEL / 32][2][32][16]; // 48 KB
  __shared__ __align__(32) unsigned short sHf[4][2][32][16];           //  8 KB
  __shared__ int   sTok[TM];
  __shared__ float sW[TM];

  const int tid  = threadIdx.x;
  const int lane = tid & 31;
  const int wv   = tid >> 5;
  const int mn   = lane & 15;
  const int kgc  = lane >> 4;                        // C/D K-group (lane[4])

  if (tid < TM) {
    int tt = t0 + tid;
    sTok[tid] = (tt < n_e) ? tlist[e * T_TOKENS + tt] : -1;
    sW[tid]   = (tt < n_e) ? twt[e * T_TOKENS + tt] : 0.0f;
  }
  __syncthreads();

  // stage x tile into A-fragment layout: kg=kk>>4, j=kk&15
  for (int i = tid; i < TM * (DMODEL / 2); i += 256) {
    int m = i / (DMODEL / 2), p = i - m * (DMODEL / 2);   // p = d/2 (d even)
    int tok = sTok[m];
    unsigned int val = 0u;
    if (tok >= 0) val = ((const unsigned int*)xbf)[(size_t)tok * (DMODEL / 2) + p];
    int d = p * 2, kt = d >> 5, kk = d & 31;
    int fl = 16 * (kk >> 4) + (m & 15);
    int j  = kk & 15;
    *(unsigned int*)&sXf[kt][m >> 4][fl][j] = val;        // j even: aligned
  }
  __syncthreads();

  v8f yacc[2][6];
#pragma unroll
  for (int mt = 0; mt < 2; ++mt)
#pragma unroll
    for (int nt = 0; nt < 6; ++nt) yacc[mt][nt] = (v8f){};

  for (int fc = 0; fc < DFF / 128; ++fc) {
    v8f gacc[2] = {(v8f){}, (v8f){}}, uacc[2] = {(v8f){}, (v8f){}};

    // ---- gate/up GEMM: K = DMODEL, wave owns 16 F-cols ----
    const unsigned short* gptr =
        wgB + ((size_t)(e * (DFF / 16) + fc * 8 + wv) * (DMODEL / 32)) * 512 + lane * 16;
    const unsigned short* uptr =
        wuB + ((size_t)(e * (DFF / 16) + fc * 8 + wv) * (DMODEL / 32)) * 512 + lane * 16;
    for (int kt = 0; kt < DMODEL / 32; ++kt) {
      Frag a0, a1, bg, bu;
      a0.u = *(const v8u*)&sXf[kt][0][lane][0];
      a1.u = *(const v8u*)&sXf[kt][1][lane][0];
      bg.u = *(const v8u*)gptr;
      bu.u = *(const v8u*)uptr;
      if ((kt & 3) == 0) {                           // prefetch 4 tiles ahead
        __builtin_prefetch((const void*)(gptr + 4 * 512), 0, 0);
        __builtin_prefetch((const void*)(uptr + 4 * 512), 0, 0);
      }
      gptr += 512; uptr += 512;
      gacc[0] = __builtin_amdgcn_wmma_f32_16x16x32_bf16(false, a0.v, false, bg.v,
                                                        (short)0, gacc[0], false, false);
      gacc[1] = __builtin_amdgcn_wmma_f32_16x16x32_bf16(false, a1.v, false, bg.v,
                                                        (short)0, gacc[1], false, false);
      uacc[0] = __builtin_amdgcn_wmma_f32_16x16x32_bf16(false, a0.v, false, bu.v,
                                                        (short)0, uacc[0], false, false);
      uacc[1] = __builtin_amdgcn_wmma_f32_16x16x32_bf16(false, a1.v, false, bu.v,
                                                        (short)0, uacc[1], false, false);
    }

    // ---- fused SwiGLU -> h in A-fragment layout ----
    __syncthreads();                                 // prior sHf readers done
#pragma unroll
    for (int mt = 0; mt < 2; ++mt)
#pragma unroll
      for (int r = 0; r < 8; ++r) {
        float g = gacc[mt][r], u = uacc[mt][r];
        float h = (g / (1.0f + __expf(-g))) * u;
        // k-in-chunk = wv*16+mn: ktile=wv>>1, kg=wv&1, j=mn
        sHf[wv >> 1][mt][16 * (wv & 1) + r + 8 * kgc][mn] =
            (unsigned short)f2bf(h);
      }
    __syncthreads();

    // ---- down GEMM: wave owns 96 D-cols (6 N-tiles) ----
    for (int k2t = 0; k2t < 4; ++k2t) {
      Frag h0, h1;
      h0.u = *(const v8u*)&sHf[k2t][0][lane][0];
      h1.u = *(const v8u*)&sHf[k2t][1][lane][0];
#pragma unroll
      for (int nt = 0; nt < 6; ++nt) {
        Frag bd;
        bd.u = *(const v8u*)(wdB +
              (((size_t)(e * (DMODEL / 16) + wv * 6 + nt) * (DFF / 32)) + fc * 4 + k2t) * 512
              + lane * 16);
        yacc[0][nt] = __builtin_amdgcn_wmma_f32_16x16x32_bf16(false, h0.v, false, bd.v,
                                                              (short)0, yacc[0][nt], false, false);
        yacc[1][nt] = __builtin_amdgcn_wmma_f32_16x16x32_bf16(false, h1.v, false, bd.v,
                                                              (short)0, yacc[1][nt], false, false);
      }
    }
  }

  // ---- scatter: out[token] += gate_weight * y (2 exact float adds/token) ----
#pragma unroll
  for (int mt = 0; mt < 2; ++mt)
#pragma unroll
    for (int r = 0; r < 8; ++r) {
      const int m   = mt * 16 + r + 8 * kgc;
      const int tok = sTok[m];
      if (tok < 0) continue;
      const float wt = sW[m];
#pragma unroll
      for (int nt = 0; nt < 6; ++nt) {
        const int d = wv * 96 + nt * 16 + mn;
        atomicAdd(&out[(size_t)tok * DMODEL + d], wt * yacc[mt][nt][r]);
      }
    }
}

// ---------------------------------------------------------------------------
extern "C" void kernel_launch(void* const* d_in, const int* in_sizes, int n_in,
                              void* d_out, int out_size, void* d_ws, size_t ws_size,
                              hipStream_t stream) {
  (void)in_sizes; (void)n_in; (void)ws_size;  // requires ws_size >= ~80 MB
  const float* x      = (const float*)d_in[0];
  const float* gate_w = (const float*)d_in[1];
  const float* w_gate = (const float*)d_in[2];
  const float* w_up   = (const float*)d_in[3];
  const float* w_down = (const float*)d_in[4];
  float* out = (float*)d_out;

  char* ws = (char*)d_ws;
  int*            counts = (int*)(ws + WS_COUNTS);
  int*            tlist  = (int*)(ws + WS_TLIST);
  float*          twt    = (float*)(ws + WS_TWT);
  unsigned short* xbf    = (unsigned short*)(ws + WS_XBF);
  unsigned short* wgB    = (unsigned short*)(ws + WS_WG);
  unsigned short* wuB    = (unsigned short*)(ws + WS_WU);
  unsigned short* wdB    = (unsigned short*)(ws + WS_WD);

  moe_init<<<(out_size + 255) / 256, 256, 0, stream>>>(out, out_size, counts);
  convert_x<<<(T_TOKENS * DMODEL / 4) / 256, 256, 0, stream>>>(x, xbf);
  convert_wgu<<<(NEXPERTS * (DMODEL / 4) * DFF) / 256, 256, 0, stream>>>(w_gate, w_up, wgB, wuB);
  convert_wd<<<(NEXPERTS * (DFF / 4) * DMODEL) / 256, 256, 0, stream>>>(w_down, wdB);
  moe_router<<<T_TOKENS / 8, 256, 0, stream>>>(x, gate_w, counts, tlist, twt);
  dim3 grid(T_TOKENS / TM, NEXPERTS);
  moe_expert_ffn<<<grid, 256, 0, stream>>>(xbf, wgB, wuB, wdB, counts, tlist, twt, out);
}
